// S4D_31396210933769
// MI455X (gfx1250) — compile-verified, matrix operand order checked by reading, not run on hardware
//
#include <hip/hip_runtime.h>
#include <hip/hip_bf16.h>
#include <math.h>

// Problem constants (from reference): B=8, H=512, L=2048, N2=32, HK=2H=1024
#define BB 8
#define HH 512
#define LL 2048
#define NN 32
#define HKK 1024

typedef __attribute__((ext_vector_type(16))) __bf16 v16bf;
typedef __attribute__((ext_vector_type(8)))  float  v8f;
typedef int gv4i __attribute__((vector_size(16)));   // matches builtin param type

#if defined(__has_builtin)
#  if __has_builtin(__builtin_amdgcn_global_load_async_to_lds_b128)
#    define HAVE_ASYNC 1
#  endif
#endif
#ifndef HAVE_ASYNC
#  define HAVE_ASYNC 0
#endif

#define AS1 __attribute__((address_space(1)))
#define AS3 __attribute__((address_space(3)))

static __device__ __forceinline__ void copy16_to_lds(const __bf16* g, __bf16* l) {
#if HAVE_ASYNC
  // CDNA5 async global->LDS DMA (tracked by ASYNCcnt)
  __builtin_amdgcn_global_load_async_to_lds_b128(
      (AS1 gv4i*)(unsigned long long)g, (AS3 gv4i*)l, 0, 0);
#else
  *(uint4*)l = *(const uint4*)g;  // fallback: through VGPRs (ds_store_b128)
#endif
}

static __device__ __forceinline__ void wait_async0() {
#if HAVE_ASYNC
#  if __has_builtin(__builtin_amdgcn_s_wait_asynccnt)
  __builtin_amdgcn_s_wait_asynccnt(0);
#  else
  asm volatile("s_wait_asynccnt 0" ::: "memory");
#  endif
#endif
}

static __device__ __forceinline__ v8f wmma_bf16(v16bf a, v16bf b, v8f c) {
  // D = A(16x32 bf16) x B(32x16 bf16) + C(16x16 f32)
  return __builtin_amdgcn_wmma_f32_16x16x32_bf16(false, a, false, b, (short)0, c,
                                                 false, false);
}

union TileU {
  uint4 q[2];
  v16bf v;
};

// ---------------------------------------------------------------------------
// 1) S4D coefficients: w = exp(dt*A), C_eff = C*(exp(dt*A)-1)/A   (HK*N2 tiny)
// ---------------------------------------------------------------------------
__global__ void coeff_kernel(const float* __restrict__ log_dt,
                             const float* __restrict__ C_ri,
                             const float* __restrict__ logAr,
                             const float* __restrict__ A_im,
                             float* __restrict__ Wr, float* __restrict__ Wi,
                             float* __restrict__ Cr, float* __restrict__ Ci) {
  int idx = blockIdx.x * blockDim.x + threadIdx.x;
  if (idx >= HKK * NN) return;
  int hk = idx / NN;
  float dt = expf(log_dt[hk]);
  float Ar = -expf(logAr[idx]);
  float Ai = A_im[idx];
  float dr = Ar * dt, di = Ai * dt;
  float e  = expf(dr);
  float wr = e * cosf(di);
  float wi = e * sinf(di);
  // q = (w - 1)/A = (w-1)*conj(A)/|A|^2
  float den = Ar * Ar + Ai * Ai;
  float xr = wr - 1.f, xi = wi;
  float qr = (xr * Ar + xi * Ai) / den;
  float qi = (xi * Ar - xr * Ai) / den;
  float cv_r = C_ri[2 * idx + 0], cv_i = C_ri[2 * idx + 1];
  Wr[idx] = wr; Wi[idx] = wi;
  Cr[idx] = cv_r * qr - cv_i * qi;
  Ci[idx] = cv_r * qi + cv_i * qr;
}

// ---------------------------------------------------------------------------
// 2) Split projection weights into bf16 hi/lo pair (near-fp32 WMMA path)
// ---------------------------------------------------------------------------
__global__ void wsplit_kernel(const float* __restrict__ w,
                              __bf16* __restrict__ whi, __bf16* __restrict__ wlo) {
  int idx = blockIdx.x * blockDim.x + threadIdx.x;
  if (idx >= 2 * HH * HH) return;
  float f = w[idx];
  __bf16 hi = (__bf16)f;
  whi[idx] = hi;
  wlo[idx] = (__bf16)(f - (float)hi);
}

// ---------------------------------------------------------------------------
// 3) Depthwise conv (k=11, pad 5, groups=H, 3 out/ch) + gate ug = v*x
//    out channel o uses input channel o/3, weight row o. z-branch unused.
// ---------------------------------------------------------------------------
__global__ void conv_gate_kernel(const float* __restrict__ u,
                                 const float* __restrict__ cw,
                                 const float* __restrict__ cb,
                                 float* __restrict__ ug) {
  size_t idx = (size_t)blockIdx.x * blockDim.x + threadIdx.x;
  if (idx >= (size_t)BB * HH * LL) return;
  int l = (int)(idx % LL);
  int h = (int)((idx / LL) % HH);
  int b = (int)(idx / ((size_t)LL * HH));
  int cx = h / 3;              // x = out channels [0,H)
  int cv = (HH + h) / 3;       // v = out channels [H,2H)
  const float* ux = u + ((size_t)b * HH + cx) * LL;
  const float* uv = u + ((size_t)b * HH + cv) * LL;
  const float* wx = cw + h * 11;
  const float* wv = cw + (HH + h) * 11;
  float xs = cb[h], vs = cb[HH + h];
#pragma unroll
  for (int t = 0; t < 11; ++t) {
    int li = l + t - 5;
    if (li >= 0 && li < LL) {
      xs = fmaf(wx[t], ux[li], xs);
      vs = fmaf(wv[t], uv[li], vs);
    }
  }
  ug[idx] = xs * vs;
}

// ---------------------------------------------------------------------------
// 4) Bidirectional diagonal-SSM scan. One wave per (b,h); lane = state n.
//    fwd:  s = w*s + ug[l];  y_f[l] = 2 Re(C.s)           (heads [0,H))
//    bwd:  y_b[l] = 2 Re(C.r); r = w*r + ug[l]            (heads [H,2H))
//    y = gelu(y_f + y_b + D*ug), stored TRANSPOSED (b,l,h) as bf16 hi/lo so
//    the GEMM's B operand becomes contiguous 16B runs.
// ---------------------------------------------------------------------------
__global__ __launch_bounds__(32) void scan_kernel(
    const float* __restrict__ ug, const float* __restrict__ Wr,
    const float* __restrict__ Wi, const float* __restrict__ Cr,
    const float* __restrict__ Ci, const float* __restrict__ D,
    float* __restrict__ yfwd, __bf16* __restrict__ yhiT, __bf16* __restrict__ yloT) {
  __shared__ float tile[32][33];
  int bh = blockIdx.x;
  int b = bh / HH, h = bh % HH;
  int lane = threadIdx.x;
  const float* ugp = ug + (size_t)bh * LL;
  float* yf = yfwd + (size_t)bh * LL;

  {  // forward scan, head h
    int c0 = h * NN + lane;
    float wr = Wr[c0], wi = Wi[c0], cr = Cr[c0], cii = Ci[c0];
    float sr = 0.f, si = 0.f;
    for (int l0 = 0; l0 < LL; l0 += 32) {
      float uc = ugp[l0 + lane];
      for (int t = 0; t < 32; ++t) {
        float uu = __shfl(uc, t, 32);
        float nsr = fmaf(wr, sr, fmaf(-wi, si, uu));
        float nsi = fmaf(wr, si, wi * sr);
        sr = nsr; si = nsi;
        tile[lane][t] = fmaf(cr, sr, -cii * si);
      }
      __syncthreads();
      float yv = 0.f;
#pragma unroll
      for (int n = 0; n < 32; ++n) yv += tile[n][lane];
      yf[l0 + lane] = 2.f * yv;
      __syncthreads();
    }
  }
  {  // backward scan, head H+h; fuse D, exact GELU, bf16 split, transpose store
    int c0 = (HH + h) * NN + lane;
    float wr = Wr[c0], wi = Wi[c0], cr = Cr[c0], cii = Ci[c0];
    float dh = D[h];
    float rr = 0.f, ri = 0.f;
    for (int l0 = LL - 32; l0 >= 0; l0 -= 32) {
      float uc = ugp[l0 + lane];
      for (int t = 31; t >= 0; --t) {
        tile[lane][t] = fmaf(cr, rr, -cii * ri);   // uses r[l] (m>l terms)
        float uu = __shfl(uc, t, 32);
        float nrr = fmaf(wr, rr, fmaf(-wi, ri, uu));
        float nri = fmaf(wr, ri, wi * rr);
        rr = nrr; ri = nri;
      }
      __syncthreads();
      float yv = 0.f;
#pragma unroll
      for (int n = 0; n < 32; ++n) yv += tile[n][lane];
      float y = yf[l0 + lane] + 2.f * yv + dh * uc;
      float g = 0.5f * y * (1.f + erff(y * 0.70710678118654752f));  // exact gelu
      __bf16 hi = (__bf16)g;
      size_t tix = ((size_t)b * LL + (l0 + lane)) * HH + h;  // (b,l,h) layout
      yhiT[tix] = hi;
      yloT[tix] = (__bf16)(g - (float)hi);
      __syncthreads();
    }
  }
}

// ---------------------------------------------------------------------------
// 5) Projection GEMM (M=1024, K=512, N=B*L) on v_wmma_f32_16x16x32_bf16 with
//    split precision (hi*hi + hi*lo + lo*hi), fused a*sigmoid(b) gate.
//    Weight K-panels are double-buffered in LDS via async global->LDS copies;
//    activations (transposed layout) load as contiguous b128 runs.
//    Wave: 16 columns x 4 row-pairs (rows m, m+512); block: 8 waves.
// ---------------------------------------------------------------------------
__global__ __launch_bounds__(256) void proj_gate_kernel(
    const __bf16* __restrict__ yhiT, const __bf16* __restrict__ yloT,
    const __bf16* __restrict__ Whi, const __bf16* __restrict__ Wlo,
    const float* __restrict__ out_b, float* __restrict__ out) {
  // Apan[buf][hi/lo][row*32+k] : rows 0..63 = m0+r ('a'), 64..127 = 512+m0+r ('b')
  __shared__ __align__(16) __bf16 Apan[2][2][128 * 32];

  int tid  = threadIdx.x;
  int lane = tid & 31;
  int wave = tid >> 5;
  int b  = blockIdx.y >> 4;
  int l0 = (blockIdx.y & 15) * 128 + wave * 16;
  int m0 = blockIdx.x * 64;
  int nc = lane & 15;            // M index (A) / N index (B) within tile
  int kb = (lane >> 4) * 8;      // K sub-base per half-wave

  // stage panel for k0 into Apan[buf]: 1024 16B-chunks, 4 per thread
  auto stage = [&](int buf, int k0) {
#pragma unroll
    for (int c = 0; c < 4; ++c) {
      int chunk = (tid << 2) | c;          // 0..1023
      int arr = chunk >> 9;                // 0=hi 1=lo
      int r   = (chunk >> 2) & 127;        // panel row
      int ko  = (chunk & 3) * 8;           // k offset (bf16 units)
      int grow = (r < 64) ? (m0 + r) : (HH + m0 + (r - 64));
      const __bf16* g = (arr ? Wlo : Whi) + (size_t)grow * HH + k0 + ko;
      copy16_to_lds(g, &Apan[buf][arr][r * 32 + ko]);
    }
  };

  v8f accA[4], accB[4];
#pragma unroll
  for (int i = 0; i < 4; ++i)
#pragma unroll
    for (int j = 0; j < 8; ++j) { accA[i][j] = 0.f; accB[i][j] = 0.f; }

  const __bf16* byh = yhiT + ((size_t)b * LL + (l0 + nc)) * HH + kb;
  const __bf16* byl = yloT + ((size_t)b * LL + (l0 + nc)) * HH + kb;

  stage(0, 0);
  wait_async0();
  __syncthreads();

  for (int k0 = 0; k0 < HH; k0 += 32) {
    int cur = (k0 >> 5) & 1;
    if (k0 + 32 < HH) stage(cur ^ 1, k0 + 32);  // prefetch next panel (async)

    // B operand from transposed activations: contiguous 16B runs
    TileU bh, bl;
    bh.q[0] = *(const uint4*)(byh + k0);
    bh.q[1] = *(const uint4*)(byh + k0 + 16);
    bl.q[0] = *(const uint4*)(byl + k0);
    bl.q[1] = *(const uint4*)(byl + k0 + 16);

#pragma unroll
    for (int mt = 0; mt < 4; ++mt) {
      int ra = mt * 16 + nc;        // 'a' half panel row
      int rb = 64 + mt * 16 + nc;   // 'b' half panel row
      TileU ah, al, ch, cl;
      const __bf16* pa = &Apan[cur][0][ra * 32 + kb];
      const __bf16* qa = &Apan[cur][1][ra * 32 + kb];
      const __bf16* pb = &Apan[cur][0][rb * 32 + kb];
      const __bf16* qb = &Apan[cur][1][rb * 32 + kb];
      ah.q[0] = *(const uint4*)pa; ah.q[1] = *(const uint4*)(pa + 16);
      al.q[0] = *(const uint4*)qa; al.q[1] = *(const uint4*)(qa + 16);
      ch.q[0] = *(const uint4*)pb; ch.q[1] = *(const uint4*)(pb + 16);
      cl.q[0] = *(const uint4*)qb; cl.q[1] = *(const uint4*)(qb + 16);
      accA[mt] = wmma_bf16(ah.v, bh.v, accA[mt]);
      accA[mt] = wmma_bf16(ah.v, bl.v, accA[mt]);
      accA[mt] = wmma_bf16(al.v, bh.v, accA[mt]);
      accB[mt] = wmma_bf16(ch.v, bh.v, accB[mt]);
      accB[mt] = wmma_bf16(ch.v, bl.v, accB[mt]);
      accB[mt] = wmma_bf16(cl.v, bh.v, accB[mt]);
    }
    wait_async0();     // my async copies into the next buffer have landed
    __syncthreads();   // everyone done computing cur + staging next
  }

  // Fused gate epilogue: out = (a + ba) * sigmoid(b + bb)
  int mo = 8 * (lane >> 4);
#pragma unroll
  for (int mt = 0; mt < 4; ++mt) {
#pragma unroll
    for (int r = 0; r < 8; ++r) {
      int o = m0 + mt * 16 + mo + r;
      float av = accA[mt][r] + out_b[o];
      float bv = accB[mt][r] + out_b[o + HH];
      out[((size_t)(b * HH + o)) * LL + l0 + nc] = av * (1.f / (1.f + expf(-bv)));
    }
  }
}

// ---------------------------------------------------------------------------
extern "C" void kernel_launch(void* const* d_in, const int* in_sizes, int n_in,
                              void* d_out, int out_size, void* d_ws, size_t ws_size,
                              hipStream_t stream) {
  const float* u      = (const float*)d_in[0];
  const float* cw     = (const float*)d_in[1];
  const float* cb     = (const float*)d_in[2];
  const float* log_dt = (const float*)d_in[3];
  const float* C_ri   = (const float*)d_in[4];
  const float* logAr  = (const float*)d_in[5];
  const float* A_im   = (const float*)d_in[6];
  const float* D      = (const float*)d_in[7];
  const float* out_w  = (const float*)d_in[8];
  const float* out_b  = (const float*)d_in[9];
  float* out = (float*)d_out;

  char* wsb = (char*)d_ws;
  size_t off = 0;
  auto alloc = [&](size_t bytes) -> char* {
    char* p = wsb + off;
    off = (off + bytes + 255) & ~(size_t)255;
    return p;
  };
  const size_t NBHL = (size_t)BB * HH * LL;  // 8,388,608
  float*  Wr   = (float*)alloc(HKK * NN * sizeof(float));
  float*  Wi   = (float*)alloc(HKK * NN * sizeof(float));
  float*  Cr   = (float*)alloc(HKK * NN * sizeof(float));
  float*  Ci   = (float*)alloc(HKK * NN * sizeof(float));
  __bf16* Whi  = (__bf16*)alloc(2 * HH * HH * sizeof(__bf16));
  __bf16* Wlo  = (__bf16*)alloc(2 * HH * HH * sizeof(__bf16));
  float*  ug   = (float*)alloc(NBHL * sizeof(float));
  float*  yfwd = (float*)alloc(NBHL * sizeof(float));
  __bf16* yhiT = (__bf16*)alloc(NBHL * sizeof(__bf16));
  __bf16* yloT = (__bf16*)alloc(NBHL * sizeof(__bf16));
  (void)ws_size; (void)in_sizes; (void)n_in; (void)out_size;

  coeff_kernel<<<(HKK * NN + 255) / 256, 256, 0, stream>>>(log_dt, C_ri, logAr,
                                                           A_im, Wr, Wi, Cr, Ci);
  wsplit_kernel<<<(2 * HH * HH + 255) / 256, 256, 0, stream>>>(out_w, Whi, Wlo);
  conv_gate_kernel<<<(int)((NBHL + 255) / 256), 256, 0, stream>>>(u, cw, cb, ug);
  scan_kernel<<<BB * HH, 32, 0, stream>>>(ug, Wr, Wi, Cr, Ci, D, yfwd, yhiT, yloT);
  dim3 g(HH / 64, BB * (LL / 128));  // (8, 128)
  proj_gate_kernel<<<g, 256, 0, stream>>>(yhiT, yloT, Whi, Wlo, out_b, out);
}
